// Attention_2576980377756
// MI455X (gfx1250) — compile-verified, matrix operand order checked by reading, not run on hardware
//
#include <hip/hip_runtime.h>
#include <hip/hip_bf16.h>

typedef __bf16 bf16_t;
typedef __attribute__((ext_vector_type(16))) __bf16 bf16x16;
typedef __attribute__((ext_vector_type(8)))  float  f32x8;

#define DIM 768
#define NHEAD 12
#define HDIM 64
#define SEQ 2048
#define MROWS 4096          // B*N
#define SCALE 0.125f        // 64^-0.5
#define MASK_VAL -10000000.0f

union Frag16 { bf16x16 v; unsigned u[8]; };

// ---- CDNA5 async global->LDS copy (ASYNCcnt path) --------------------------
__device__ __forceinline__ unsigned lds_off(const void* p) {
  // generic pointer to LDS = {shared_aperture_hi, lds_byte_offset}
  return (unsigned)(uintptr_t)p;
}
__device__ __forceinline__ void async_load_b128(unsigned lds_addr,
                                                const void* gbase,
                                                unsigned goff_bytes) {
  asm volatile("global_load_async_to_lds_b128 %0, %1, %2 offset:0"
               :: "v"(lds_addr), "v"(goff_bytes), "s"(gbase)
               : "memory");
}
__device__ __forceinline__ void wait_async0() {
  asm volatile("s_wait_asynccnt 0" ::: "memory");
}

// Load a 16x32 bf16 fragment (A or B operand) from a row-major LDS tile.
// ISA 7.12.2 16-bit layout: lane m = lane&15 holds row m; half = lane>>4;
// VGPR v (v<4): K = 2v,2v+1 (+8*half); VGPR v (v>=4): K = 16+2(v-4),+1 (+8*half).
__device__ __forceinline__ bf16x16 load_frag16(const bf16_t* __restrict__ base,
                                               int rowOff, int stride, int kOff) {
  const int lane = threadIdx.x & 31;
  const int half = lane >> 4;
  const int m = lane & 15;
  const bf16_t* p = base + (rowOff + m) * stride + kOff + half * 8;
  Frag16 f;
#pragma unroll
  for (int v = 0; v < 8; ++v) {
    const int kb = ((v & 3) * 2) + ((v >> 2) * 16);
    f.u[v] = *(const unsigned*)(p + kb);
  }
  return f.v;
}

__device__ __forceinline__ f32x8 wmma_bf16(bf16x16 a, bf16x16 b, f32x8 c) {
  return __builtin_amdgcn_wmma_f32_16x16x32_bf16(false, a, false, b,
                                                 (short)0, c, false, false);
}

// ---------------------------------------------------------------- cast f32->bf16
__global__ void cast_f32_to_bf16_x4(const float* __restrict__ s,
                                    bf16_t* __restrict__ d, int n4) {
  int i = blockIdx.x * blockDim.x + threadIdx.x;
  if (i < n4) {
    float4 f = ((const float4*)s)[i];
    bf16_t* p = d + i * 4;
    p[0] = (bf16_t)f.x; p[1] = (bf16_t)f.y;
    p[2] = (bf16_t)f.z; p[3] = (bf16_t)f.w;
  }
}

// ---------------------------------------------------------------- 128x128 GEMM
// out[row, col] = sum_k A[row,k] * W[col,k]      (A: [M,768], W: [Nout,768])
// MODE 0: scatter bf16 into q/k/v [B,H,N,64]. MODE 1: f32 out + bias.
// Double-buffered async global->LDS staging; LDS pitch 40 (80B: 16B-aligned
// rows for async B128, 20-bank stride -> conflict-free fragment reads).
template <int MODE>
__global__ __launch_bounds__(256) void gemm128(
    const bf16_t* __restrict__ A, const bf16_t* __restrict__ W,
    bf16_t* __restrict__ qb, bf16_t* __restrict__ kb_, bf16_t* __restrict__ vb,
    const float* __restrict__ bias, float* __restrict__ out, int Nout) {
  __shared__ bf16_t As[2][128][40];
  __shared__ bf16_t Ws[2][128][40];
  const int t = threadIdx.x;
  const int waveId = t >> 5;
  const int lane = t & 31;
  const int half = lane >> 4;
  const int n16 = lane & 15;
  const int mBase = blockIdx.y * 128;
  const int nBase = blockIdx.x * 128;
  const int wm = (waveId & 3) * 32;   // 4 waves along M
  const int wn = (waveId >> 2) * 64;  // 2 waves along N

  // each thread stages 2 x 16B for A and 2 x 16B for W per K-step
  auto stage = [&](int buf, int k0) {
#pragma unroll
    for (int i = 0; i < 2; ++i) {
      const int c = i * 256 + t;          // 0..511
      const int row = c >> 2;             // 0..127
      const int q = (c & 3) * 16;         // byte offset within 64B row slice
      async_load_b128(lds_off(&As[buf][row][0]) + q, A,
                      (unsigned)((mBase + row) * (DIM * 2) + k0 * 2 + q));
      async_load_b128(lds_off(&Ws[buf][row][0]) + q, W,
                      (unsigned)((nBase + row) * (DIM * 2) + k0 * 2 + q));
    }
  };

  f32x8 acc[2][4] = {};
  stage(0, 0);

  for (int k0 = 0, it = 0; k0 < DIM; k0 += 32, ++it) {
    const int buf = it & 1;
    wait_async0();        // own async copies for `buf` are in LDS
    __syncthreads();      // everyone's are; everyone done reading buf^1
    if (k0 + 32 < DIM) stage(buf ^ 1, k0 + 32);

    bf16x16 af[2], bf_[4];
#pragma unroll
    for (int i = 0; i < 2; ++i)
      af[i] = load_frag16(&As[buf][0][0], wm + i * 16, 40, 0);
#pragma unroll
    for (int j = 0; j < 4; ++j)
      bf_[j] = load_frag16(&Ws[buf][0][0], wn + j * 16, 40, 0);
#pragma unroll
    for (int i = 0; i < 2; ++i)
#pragma unroll
      for (int j = 0; j < 4; ++j) acc[i][j] = wmma_bf16(af[i], bf_[j], acc[i][j]);
  }

#pragma unroll
  for (int i = 0; i < 2; ++i)
#pragma unroll
    for (int j = 0; j < 4; ++j)
#pragma unroll
      for (int r = 0; r < 8; ++r) {
        const int row = mBase + wm + i * 16 + r + half * 8;
        const int col = nBase + wn + j * 16 + n16;
        const float val = acc[i][j][r];
        if (MODE == 0) {
          const int which = col / DIM;
          const int rr = col % DIM;
          const int h = rr >> 6, hd = rr & 63;
          const int b = row >> 11, n = row & 2047;
          bf16_t* dst = (which == 0) ? qb : (which == 1) ? kb_ : vb;
          dst[(((size_t)(b * NHEAD + h) << 11) + n) * HDIM + hd] = (bf16_t)val;
        } else {
          out[(size_t)row * Nout + col] = val + bias[col];
        }
      }
}

// ---------------------------------------------------------------- flash attention
// grid: (B*H*16) blocks; each block = 128 query rows; each wave = 16 rows.
__global__ __launch_bounds__(256) void attn_kernel(
    const bf16_t* __restrict__ Q, const bf16_t* __restrict__ K,
    const bf16_t* __restrict__ V, const int* __restrict__ pm,
    bf16_t* __restrict__ out) {
  __shared__ bf16_t Ks[64][72];       // [kpos][hd], pitch 144B (16B-aligned)
  __shared__ bf16_t Vt[64][72];       // transposed: [hd][kpos]
  __shared__ bf16_t Ps[8][16][72];    // per-wave P scratch
  __shared__ unsigned char maskc[SEQ];

  const int t = threadIdx.x;
  const int waveId = t >> 5;
  const int lane = t & 31;
  const int half = lane >> 4;
  const int n16 = lane & 15;

  const int blk = blockIdx.x;
  const int qTile = blk & 15;
  const int bh = blk >> 4;            // 0..23
  const int b = bh / NHEAD;
  const int h = bh % NHEAD;
  const bf16_t* Qp = Q + ((size_t)bh << 11) * HDIM;
  const bf16_t* Kp = K + ((size_t)bh << 11) * HDIM;
  const bf16_t* Vp = V + ((size_t)bh << 11) * HDIM;

#pragma unroll
  for (int i = 0; i < 8; ++i) {
    int n = i * 256 + t;
    maskc[n] = (unsigned char)(pm[(b << 11) + n] > 0);
  }

  const int qRow = qTile * 128 + waveId * 16;
  bf16x16 qf[2];
  qf[0] = load_frag16(Qp, qRow, HDIM, 0);
  qf[1] = load_frag16(Qp, qRow, HDIM, 32);

  float m_i[8], l_i[8];
#pragma unroll
  for (int r = 0; r < 8; ++r) { m_i[r] = -3.0e38f; l_i[r] = 0.0f; }
  f32x8 o[4] = {};

  for (int kc = 0; kc < SEQ; kc += 64) {
    __syncthreads();   // all waves done reading Ks/Vt (WMMA forced dscnt waits)

    // async-stage K chunk [64][64]: 512 x 16B, 2 per thread
#pragma unroll
    for (int i = 0; i < 2; ++i) {
      const int c = i * 256 + t;
      const int r = c >> 3, q = (c & 7) * 16;
      async_load_b128(lds_off(&Ks[r][0]) + q, Kp,
                      (unsigned)((kc + r) * (HDIM * 2) + q));
    }
    // stage V chunk transposed: each thread reads 16 contiguous bf16 (2x16B)
    {
      const int r = t >> 2, cq = (t & 3) * 16;
      union { uint4 u[2]; bf16_t hv[16]; } tmp;
      const uint4* vp = (const uint4*)&Vp[(size_t)(kc + r) * HDIM + cq];
      tmp.u[0] = vp[0];
      tmp.u[1] = vp[1];
#pragma unroll
      for (int j = 0; j < 16; ++j) Vt[cq + j][r] = tmp.hv[j];
    }
    wait_async0();
    __syncthreads();

    // S = Q Kchunk^T : 4 n-tiles of 16 key positions
    f32x8 s[4] = {};
#pragma unroll
    for (int j = 0; j < 4; ++j) {
      bf16x16 b0 = load_frag16(&Ks[0][0], j * 16, 72, 0);
      bf16x16 b1 = load_frag16(&Ks[0][0], j * 16, 72, 32);
      s[j] = wmma_bf16(qf[0], b0, s[j]);
      s[j] = wmma_bf16(qf[1], b1, s[j]);
    }

    // scale + padding mask (mask>0 -> MASK_VAL)
#pragma unroll
    for (int j = 0; j < 4; ++j) {
      const int ncol = kc + j * 16 + n16;
      const bool msk = maskc[ncol] != 0;
#pragma unroll
      for (int r = 0; r < 8; ++r)
        s[j][r] = msk ? MASK_VAL : s[j][r] * SCALE;
    }

    // row max across the 64-wide chunk (lane-local over j, then 16-lane group)
    float tmax[8];
#pragma unroll
    for (int r = 0; r < 8; ++r)
      tmax[r] = fmaxf(fmaxf(s[0][r], s[1][r]), fmaxf(s[2][r], s[3][r]));
#pragma unroll
    for (int xm = 1; xm < 16; xm <<= 1)
#pragma unroll
      for (int r = 0; r < 8; ++r)
        tmax[r] = fmaxf(tmax[r], __shfl_xor(tmax[r], xm, 32));

    float alpha[8];
#pragma unroll
    for (int r = 0; r < 8; ++r) {
      const float mn = fmaxf(m_i[r], tmax[r]);
      alpha[r] = __expf(m_i[r] - mn);
      m_i[r] = mn;
      l_i[r] *= alpha[r];
    }

    // P = exp(S - m); lane-partial row sums into l; P -> per-wave LDS (bf16)
#pragma unroll
    for (int j = 0; j < 4; ++j)
#pragma unroll
      for (int r = 0; r < 8; ++r) {
        const float p = __expf(s[j][r] - m_i[r]);
        l_i[r] += p;
        Ps[waveId][r + half * 8][j * 16 + n16] = (bf16_t)p;
      }

#pragma unroll
    for (int j = 0; j < 4; ++j)
#pragma unroll
      for (int r = 0; r < 8; ++r) o[j][r] *= alpha[r];

    // O += P @ Vchunk  (P reloaded as A fragments from LDS)
    bf16x16 pa0 = load_frag16(&Ps[waveId][0][0], 0, 72, 0);
    bf16x16 pa1 = load_frag16(&Ps[waveId][0][0], 0, 72, 32);
#pragma unroll
    for (int j = 0; j < 4; ++j) {
      bf16x16 vb0 = load_frag16(&Vt[0][0], j * 16, 72, 0);
      bf16x16 vb1 = load_frag16(&Vt[0][0], j * 16, 72, 32);
      o[j] = wmma_bf16(pa0, vb0, o[j]);
      o[j] = wmma_bf16(pa1, vb1, o[j]);
    }
  }

  // final l = sum over the 16-lane column group
#pragma unroll
  for (int xm = 1; xm < 16; xm <<= 1)
#pragma unroll
    for (int r = 0; r < 8; ++r) l_i[r] += __shfl_xor(l_i[r], xm, 32);

  // write O/l as bf16 to [B, N, DIM] (head-concat layout) for the FC GEMM
#pragma unroll
  for (int j = 0; j < 4; ++j)
#pragma unroll
    for (int r = 0; r < 8; ++r) {
      const int n = qRow + r + half * 8;
      const int col = h * HDIM + j * 16 + n16;
      out[((size_t)b * SEQ + n) * DIM + col] = (bf16_t)(o[j][r] / l_i[r]);
    }
}

// ---------------------------------------------------------------- launcher
extern "C" void kernel_launch(void* const* d_in, const int* in_sizes, int n_in,
                              void* d_out, int out_size, void* d_ws, size_t ws_size,
                              hipStream_t stream) {
  const float* x     = (const float*)d_in[0];   // [2,2048,768]
  const int*   pm    = (const int*)d_in[1];     // [2,2048]
  const float* w_qkv = (const float*)d_in[2];   // [2304,768]
  const float* w_fc  = (const float*)d_in[3];   // [768,768]
  const float* b_fc  = (const float*)d_in[4];   // [768]
  float* out = (float*)d_out;

  char* ws = (char*)d_ws;
  const size_t SZ_X    = (size_t)MROWS * DIM * 2;
  const size_t SZ_WQKV = (size_t)3 * DIM * DIM * 2;
  const size_t SZ_WFC  = (size_t)DIM * DIM * 2;
  const size_t SZ_HEAD = (size_t)2 * NHEAD * SEQ * HDIM * 2;

  bf16_t* xb    = (bf16_t*)(ws);
  bf16_t* wqkvb = (bf16_t*)(ws + SZ_X);
  bf16_t* wfcb  = (bf16_t*)(ws + SZ_X + SZ_WQKV);
  bf16_t* qb    = (bf16_t*)(ws + SZ_X + SZ_WQKV + SZ_WFC);
  bf16_t* kb    = (bf16_t*)(ws + SZ_X + SZ_WQKV + SZ_WFC + SZ_HEAD);
  bf16_t* vb    = (bf16_t*)(ws + SZ_X + SZ_WQKV + SZ_WFC + 2 * SZ_HEAD);
  bf16_t* attnb = (bf16_t*)(ws + SZ_X + SZ_WQKV + SZ_WFC + 3 * SZ_HEAD);

  {
    int n4 = MROWS * DIM / 4;
    cast_f32_to_bf16_x4<<<(n4 + 255) / 256, 256, 0, stream>>>(x, xb, n4);
  }
  {
    int n4 = 3 * DIM * DIM / 4;
    cast_f32_to_bf16_x4<<<(n4 + 255) / 256, 256, 0, stream>>>(w_qkv, wqkvb, n4);
  }
  {
    int n4 = DIM * DIM / 4;
    cast_f32_to_bf16_x4<<<(n4 + 255) / 256, 256, 0, stream>>>(w_fc, wfcb, n4);
  }

  gemm128<0><<<dim3(3 * DIM / 128, MROWS / 128), 256, 0, stream>>>(
      xb, wqkvb, qb, kb, vb, nullptr, nullptr, 3 * DIM);

  attn_kernel<<<2 * NHEAD * (SEQ / 128), 256, 0, stream>>>(qb, kb, vb, pm, attnb);

  gemm128<1><<<dim3(DIM / 128, MROWS / 128), 256, 0, stream>>>(
      attnb, wfcb, nullptr, nullptr, nullptr, b_fc, out, DIM);
}
